// EMAModule_45638322487447
// MI455X (gfx1250) — matrix operand order repaired and checked by reading, not compiled, not run on hardware
//
#include <hip/hip_runtime.h>
#include <hip/hip_bf16.h>

typedef __bf16 bf16_t;
typedef __attribute__((ext_vector_type(16))) __bf16 v16bf;
typedef __attribute__((ext_vector_type(8)))  __bf16 v8bf;
typedef __attribute__((ext_vector_type(8)))  float  v8f;
typedef __attribute__((ext_vector_type(4)))  unsigned int u32x4;
typedef __attribute__((ext_vector_type(8)))  int i32x8;
typedef __attribute__((ext_vector_type(4)))  int i32x4;

#define LDSTR 40   // padded LDS row stride (32 elems + 8 pad) -> conflict-free b128 reads

constexpr int Bc = 16, Cc = 512, Nn = 4096, Kc = 64;

static __device__ __forceinline__ v8f wmma_bf16(v16bf a, v16bf b, v8f c) {
  return __builtin_amdgcn_wmma_f32_16x16x32_bf16(false, a, false, b, (short)0, c, false, false);
}

// A-fragment: 16-bit A 16x32 layout. Lane(l,h): row m=l,
// elems 0..7 = K in [8h, 8h+7], elems 8..15 = K in [16+8h, 16+8h+7].
static __device__ __forceinline__ v16bf load_a(const bf16_t* row, int h) {
  union { v16bf v; v8bf p[2]; } u;
  u.p[0] = *reinterpret_cast<const v8bf*>(row + 8 * h);
  u.p[1] = *reinterpret_cast<const v8bf*>(row + 16 + 8 * h);
  return u.v;
}
// B-fragment (from transposed LDS tile Bt[n][k]): lane(l,h): col n=l,
// elems 0..7 = K in [8h, 8h+7], elems 8..15 = K in [16+8h, 16+8h+7].
static __device__ __forceinline__ v16bf load_b(const bf16_t* row, int h) {
  union { v16bf v; v8bf p[2]; } u;
  u.p[0] = *reinterpret_cast<const v8bf*>(row + 8 * h);
  u.p[1] = *reinterpret_cast<const v8bf*>(row + 16 + 8 * h);
  return u.v;
}

static __device__ __forceinline__ void wave_mma_step(const bf16_t* At, const bf16_t* Bt,
                                                     int wm, int wn, int lane,
                                                     v8f (&acc)[2][2]) {
  const int h = lane >> 4, l = lane & 15;
  v16bf a0 = load_a(At + (size_t)(wm + l) * LDSTR, h);
  v16bf a1 = load_a(At + (size_t)(wm + 16 + l) * LDSTR, h);
  v16bf b0 = load_b(Bt + (size_t)(wn + l) * LDSTR, h);
  v16bf b1 = load_b(Bt + (size_t)(wn + 16 + l) * LDSTR, h);
  acc[0][0] = wmma_bf16(a0, b0, acc[0][0]);
  acc[0][1] = wmma_bf16(a0, b1, acc[0][1]);
  acc[1][0] = wmma_bf16(a1, b0, acc[1][0]);
  acc[1][1] = wmma_bf16(a1, b1, acc[1][1]);
}

static __device__ __forceinline__ void zero_acc(v8f (&acc)[2][2]) {
  v8f z = {0.f, 0.f, 0.f, 0.f, 0.f, 0.f, 0.f, 0.f};
  acc[0][0] = z; acc[0][1] = z; acc[1][0] = z; acc[1][1] = z;
}

// ---- CDNA5 async global->LDS copy (ASYNCcnt path), 16B per lane --------------------
static __device__ __forceinline__ void async_copy_b128(unsigned int lds_off,
                                                       const void* gptr) {
  asm volatile("global_load_async_to_lds_b128 %0, %1, off"
               :: "v"(lds_off), "v"((unsigned long long)(uintptr_t)gptr)
               : "memory");
}
static __device__ __forceinline__ void wait_asynccnt0() {
  asm volatile("s_wait_asynccnt 0x0" ::: "memory");
}

// ---- CDNA5 Tensor Data Mover: 2D tile global->LDS with HW row padding --------------
// Loads rows x (32*2B) tile, row stride 512 bf16, into LDS at lds_off with 16B of
// padding after every 64B row chunk (-> 80B LDS row stride == LDSTR).
static __device__ __forceinline__ void tdm_load_tile_2d(unsigned int lds_off,
                                                        const bf16_t* gsrc,
                                                        int rows) {
  unsigned long long ga = (unsigned long long)(uintptr_t)gsrc;
  u32x4 g0;
  g0[0] = 1u;                                   // count=1 (valid user descriptor)
  g0[1] = lds_off;                              // lds_addr
  g0[2] = (unsigned int)ga;                     // global_addr[31:0]
  g0[3] = (unsigned int)((ga >> 32) & 0x01FFFFFFu) | (2u << 30);  // ga[56:32] | type=2
  i32x8 g1;
  // data_size=1(2B)<<16 | pad_enable<<20 | pad_interval=3(16 dw)<<22 | pad_amount=3(4 dw)<<25
  g1[0] = (int)((1u << 16) | (1u << 20) | (3u << 22) | (3u << 25));
  g1[1] = (int)(512u << 16);                    // tensor_dim0[15:0]=512 in bits 63:48
  g1[2] = (int)((unsigned)rows << 16);          // tensor_dim1[15:0]=rows in bits 95:80
  g1[3] = (int)(32u << 16);                     // tile_dim0=32 in bits 127:112
  g1[4] = rows;                                 // tile_dim1=rows in bits 143:128
  g1[5] = 512;                                  // tensor_dim0_stride[31:0] (elems)
  g1[6] = 0;
  g1[7] = 0;
  i32x4 g2 = {0, 0, 0, 0};
  i32x4 g3 = {0, 0, 0, 0};
  i32x8 g4 = {0, 0, 0, 0, 0, 0, 0, 0};
  __builtin_amdgcn_tensor_load_to_lds(g0, g1, g2, g3, g4, 0);
}

// ---------------- prep: weight transposes -> bf16, mu init, BN constants ------------
__global__ __launch_bounds__(256) void k_prep(
    const float* __restrict__ w_in, const float* __restrict__ w_out,
    const float* __restrict__ bases, const float* __restrict__ gamma,
    const float* __restrict__ beta, const float* __restrict__ rmean,
    const float* __restrict__ rvar,
    bf16_t* __restrict__ winT, bf16_t* __restrict__ woutT,
    bf16_t* __restrict__ mu, float* __restrict__ bns, float* __restrict__ bnb) {
  int idx = blockIdx.x * 256 + threadIdx.x;
  if (idx < Cc * Cc) {                 // transposed: wT[c][o] = w[o][c]
    int c = idx >> 9, o = idx & 511;
    winT[idx]  = (bf16_t)w_in[o * Cc + c];
    woutT[idx] = (bf16_t)w_out[o * Cc + c];
  }
  if (idx < Bc * Kc * Cc) {            // mu init = bases, replicated per batch
    int kc = idx & (Kc * Cc - 1);
    mu[idx] = (bf16_t)bases[kc];
  }
  if (idx < Cc) {
    float inv = gamma[idx] * rsqrtf(rvar[idx] + 1e-5f);
    bns[idx] = inv;
    bnb[idx] = beta[idx] - rmean[idx] * inv;
  }
}

// ---------------- conv_in: xf[b,n,o] = sum_c x[b,c,n] * w_in[o,c] + b_in[o] ---------
__global__ __launch_bounds__(256) void k_conv_in(
    const float* __restrict__ x, const bf16_t* __restrict__ winT,
    const float* __restrict__ b_in, bf16_t* __restrict__ xf) {
  const int n0 = blockIdx.x * 128, o0 = blockIdx.y * 64, b = blockIdx.z;
  __shared__ alignas(16) unsigned char smem[15360];
  bf16_t* At = (bf16_t*)smem;              // [128][LDSTR]
  bf16_t* Bt = (bf16_t*)(smem + 10240);    // [64][LDSTR]
  const int t = threadIdx.x, lane = t & 31, w = t >> 5;
  const int wm = (w >> 1) * 32, wn = (w & 1) * 32;
  v8f acc[2][2]; zero_acc(acc);
  const size_t xb = (size_t)b * Cc * Nn;
  for (int kt = 0; kt < Cc; kt += 32) {
    { // A[m][c'] = x[b][kt+c'][n0+m]  (fp32 -> bf16, transposed into LDS)
      int cp = t >> 3, m0 = (t & 7) * 16;
      const float* src = x + xb + (size_t)(kt + cp) * Nn + n0 + m0;
      bf16_t* dst = At + (size_t)m0 * LDSTR + cp;
#pragma unroll
      for (int i = 0; i < 16; i += 4) {
        float4 v = *reinterpret_cast<const float4*>(src + i);
        dst[(i + 0) * LDSTR] = (bf16_t)v.x;
        dst[(i + 1) * LDSTR] = (bf16_t)v.y;
        dst[(i + 2) * LDSTR] = (bf16_t)v.z;
        dst[(i + 3) * LDSTR] = (bf16_t)v.w;
      }
    }
    { // Bt[o'][c'] = winT[(kt+c')*512 + o0+o']
#pragma unroll
      for (int r = 0; r < 8; ++r) {
        int flat = t + r * 256;
        int i = flat >> 6, j = flat & 63;
        Bt[(size_t)j * LDSTR + i] = winT[(size_t)(kt + i) * Cc + o0 + j];
      }
    }
    __syncthreads();
    wave_mma_step(At, Bt, wm, wn, lane, acc);
    __syncthreads();
  }
  const int h = lane >> 4, l = lane & 15;
#pragma unroll
  for (int i = 0; i < 2; ++i)
#pragma unroll
    for (int j = 0; j < 2; ++j) {
      int go = o0 + wn + j * 16 + l;
      float bias = b_in[go];
#pragma unroll
      for (int r = 0; r < 8; ++r) {
        int gm = n0 + wm + i * 16 + r + 8 * h;
        xf[((size_t)b * Nn + gm) * Cc + go] = (bf16_t)(acc[i][j][r] + bias);
      }
    }
}

// ---------------- z = softmax_k( xf @ mu^T ), fused ---------------------------------
// A-tile staged via ASYNC global->LDS copies; B-tile (mu) staged via the TDM with
// hardware row padding producing the LDSTR layout directly.
__global__ __launch_bounds__(256) void k_z(
    const bf16_t* __restrict__ xf, const bf16_t* __restrict__ mu,
    bf16_t* __restrict__ zb) {
  const int n0 = blockIdx.x * 128, b = blockIdx.z;
  __shared__ alignas(16) unsigned char smem[33792];
  bf16_t* At = (bf16_t*)smem;
  bf16_t* Bt = (bf16_t*)(smem + 10240);
  const unsigned int at_off = (unsigned int)(uintptr_t)At;
  const unsigned int bt_off = (unsigned int)(uintptr_t)Bt;
  const int t = threadIdx.x, lane = t & 31, w = t >> 5;
  const int wm = (w >> 1) * 32, wn = (w & 1) * 32;
  const int row = t >> 1, part = t & 1;
  v8f acc[2][2]; zero_acc(acc);
  for (int kt = 0; kt < Cc; kt += 32) {
    // A rows from xf (row-major): one async b128 per thread
    async_copy_b128(at_off + (unsigned)(row * 80 + part * 16),
                    xf + ((size_t)(b * Nn + n0 + row)) * Cc + kt + part * 16);
    if (w == 0) {  // Bt[kk][c'] = mu[b,kk,kt+c'] : 64x32 bf16 tile via TDM
      tdm_load_tile_2d(bt_off, mu + ((size_t)b * Kc) * Cc + kt, Kc);
    }
    wait_asynccnt0();
    if (w == 0) __builtin_amdgcn_s_wait_tensorcnt(0);
    __syncthreads();
    wave_mma_step(At, Bt, wm, wn, lane, acc);
    __syncthreads();
  }
  // dump logits (fp32) to LDS, then row softmax (LBDA = 1.0)
  float* L = (float*)smem;   // [128][66]
  const int h = lane >> 4, l = lane & 15;
#pragma unroll
  for (int i = 0; i < 2; ++i)
#pragma unroll
    for (int j = 0; j < 2; ++j)
#pragma unroll
      for (int r = 0; r < 8; ++r)
        L[(size_t)(wm + i * 16 + r + 8 * h) * 66 + wn + j * 16 + l] = acc[i][j][r];
  __syncthreads();
  if (t < 128) {
    float* rowp = L + (size_t)t * 66;
    float mx = -3.4e38f;
#pragma unroll 8
    for (int k = 0; k < Kc; ++k) mx = fmaxf(mx, rowp[k]);
    float s = 0.f;
#pragma unroll 8
    for (int k = 0; k < Kc; ++k) { float e = __expf(rowp[k] - mx); rowp[k] = e; s += e; }
    float inv = 1.f / s;
    bf16_t* zr = zb + ((size_t)(b * Nn + n0 + t)) * Kc;
#pragma unroll 8
    for (int k = 0; k < Kc; ++k) zr[k] = (bf16_t)(rowp[k] * inv);
  }
}

// ---------------- ztx[b,k,c] = sum_n z[b,n,k] * xf[b,n,c] ---------------------------
__global__ __launch_bounds__(256) void k_mu_gemm(
    const bf16_t* __restrict__ zb, const bf16_t* __restrict__ xf,
    float* __restrict__ ztx) {
  const int c0 = blockIdx.x * 128, b = blockIdx.z;
  __shared__ alignas(16) unsigned char smem[15360];
  bf16_t* At = (bf16_t*)smem;              // [64][LDSTR]  (m = k of mu)
  bf16_t* Bt = (bf16_t*)(smem + 5120);     // [128][LDSTR] (n_out = c)
  const int t = threadIdx.x, lane = t & 31, w = t >> 5;
  const int wm = (w >> 2) * 32, wn = (w & 3) * 32;
  v8f acc[2][2]; zero_acc(acc);
  for (int nt = 0; nt < Nn; nt += 32) {
    { // At[k][nd] = z[b, nt+nd, k]   (z row contiguous over k -> transpose into LDS)
      int nd = t >> 3, seg = t & 7;
      v8bf v = *reinterpret_cast<const v8bf*>(
          zb + ((size_t)(b * Nn + nt + nd)) * Kc + seg * 8);
#pragma unroll
      for (int q = 0; q < 8; ++q) At[(size_t)(seg * 8 + q) * LDSTR + nd] = v[q];
    }
    { // Bt[c][nd] = xf[b, nt+nd, c0+c]
      int nd = t >> 3, seg = t & 7;
      const bf16_t* src = xf + ((size_t)(b * Nn + nt + nd)) * Cc + c0 + seg * 16;
      v8bf v0 = *reinterpret_cast<const v8bf*>(src);
      v8bf v1 = *reinterpret_cast<const v8bf*>(src + 8);
#pragma unroll
      for (int q = 0; q < 8; ++q) {
        Bt[(size_t)(seg * 16 + q) * LDSTR + nd] = v0[q];
        Bt[(size_t)(seg * 16 + 8 + q) * LDSTR + nd] = v1[q];
      }
    }
    __syncthreads();
    wave_mma_step(At, Bt, wm, wn, lane, acc);
    __syncthreads();
  }
  const int h = lane >> 4, l = lane & 15;
#pragma unroll
  for (int i = 0; i < 2; ++i)
#pragma unroll
    for (int j = 0; j < 2; ++j)
#pragma unroll
      for (int r = 0; r < 8; ++r) {
        int gk = wm + i * 16 + r + 8 * h;
        int gc = c0 + wn + j * 16 + l;
        ztx[((size_t)(b * Kc + gk)) * Cc + gc] = acc[i][j][r];
      }
}

// ---------------- mu = normalize_rows(ztx)  (colsum division cancels in L2 norm) ----
__global__ __launch_bounds__(256) void k_mu_norm(
    const float* __restrict__ ztx, bf16_t* __restrict__ mu) {
  const int w = threadIdx.x >> 5, lane = threadIdx.x & 31;
  const int row = blockIdx.x * 8 + w;          // 0 .. B*K-1
  const float* src = ztx + (size_t)row * Cc;
  float ss = 0.f;
  for (int c = lane; c < Cc; c += 32) { float v = src[c]; ss += v * v; }
#pragma unroll
  for (int off = 16; off > 0; off >>= 1) ss += __shfl_xor(ss, off, 32);
  float scale = 1.f / fmaxf(sqrtf(ss), 1e-12f);
  bf16_t* dst = mu + (size_t)row * Cc;
  for (int c = lane; c < Cc; c += 32) dst[c] = (bf16_t)(src[c] * scale);
}

// ---------------- recon[b,n,c] = sum_k z[b,n,k] * mu[b,k,c] -------------------------
__global__ __launch_bounds__(256) void k_recon(
    const bf16_t* __restrict__ zb, const bf16_t* __restrict__ mu,
    bf16_t* __restrict__ rec) {
  const int n0 = blockIdx.x * 128, c0 = blockIdx.y * 64, b = blockIdx.z;
  __shared__ alignas(16) unsigned char smem[15360];
  bf16_t* At = (bf16_t*)smem;
  bf16_t* Bt = (bf16_t*)(smem + 10240);
  const unsigned int at_off = (unsigned int)(uintptr_t)At;
  const int t = threadIdx.x, lane = t & 31, w = t >> 5;
  const int wm = (w >> 1) * 32, wn = (w & 1) * 32;
  const int row = t >> 1, part = t & 1;
  v8f acc[2][2]; zero_acc(acc);
  for (int kt = 0; kt < Kc; kt += 32) {
    // A rows from z (row-major over k): async copy
    async_copy_b128(at_off + (unsigned)(row * 80 + part * 16),
                    zb + ((size_t)(b * Nn + n0 + row)) * Kc + kt + part * 16);
    { // Bt[c][kd] = mu[b, kt+kd, c0+c]  (transpose into LDS)
      int kd = t >> 3, seg = t & 7;
      v8bf v = *reinterpret_cast<const v8bf*>(
          mu + ((size_t)(b * Kc + kt + kd)) * Cc + c0 + seg * 8);
#pragma unroll
      for (int q = 0; q < 8; ++q) Bt[(size_t)(seg * 8 + q) * LDSTR + kd] = v[q];
    }
    wait_asynccnt0();
    __syncthreads();
    wave_mma_step(At, Bt, wm, wn, lane, acc);
    __syncthreads();
  }
  const int h = lane >> 4, l = lane & 15;
#pragma unroll
  for (int i = 0; i < 2; ++i)
#pragma unroll
    for (int j = 0; j < 2; ++j)
#pragma unroll
      for (int r = 0; r < 8; ++r) {
        int gm = n0 + wm + i * 16 + r + 8 * h;
        int gc = c0 + wn + j * 16 + l;
        rec[((size_t)b * Nn + gm) * Cc + gc] = (bf16_t)acc[i][j][r];
      }
}

// ---------------- conv_out + BN(eval) + residual, (B,C,N) coalesced output ----------
__global__ __launch_bounds__(256) void k_conv_out(
    const bf16_t* __restrict__ rec, const bf16_t* __restrict__ woutT,
    const float* __restrict__ b_out, const float* __restrict__ bns,
    const float* __restrict__ bnb, const float* __restrict__ x,
    float* __restrict__ out) {
  const int n0 = blockIdx.x * 128, o0 = blockIdx.y * 64, b = blockIdx.z;
  __shared__ alignas(16) unsigned char smem[33792];
  bf16_t* At = (bf16_t*)smem;
  bf16_t* Bt = (bf16_t*)(smem + 10240);
  const unsigned int at_off = (unsigned int)(uintptr_t)At;
  const int t = threadIdx.x, lane = t & 31, w = t >> 5;
  const int wm = (w >> 1) * 32, wn = (w & 1) * 32;
  const int row = t >> 1, part = t & 1;
  v8f acc[2][2]; zero_acc(acc);
  for (int kt = 0; kt < Cc; kt += 32) {
    // A rows from rec (row-major over c): async copy
    async_copy_b128(at_off + (unsigned)(row * 80 + part * 16),
                    rec + ((size_t)(b * Nn + n0 + row)) * Cc + kt + part * 16);
    { // Bt[o'][c'] = woutT[(kt+c')*512 + o0+o']
#pragma unroll
      for (int r = 0; r < 8; ++r) {
        int flat = t + r * 256;
        int i = flat >> 6, j = flat & 63;
        Bt[(size_t)j * LDSTR + i] = woutT[(size_t)(kt + i) * Cc + o0 + j];
      }
    }
    wait_asynccnt0();
    __syncthreads();
    wave_mma_step(At, Bt, wm, wn, lane, acc);
    __syncthreads();
  }
  // transpose fp32 tile through LDS -> coalesced (B,C,N) stores with BN + residual
  float* L = (float*)smem;   // [64][132]  (o rows, m cols)
  const int h = lane >> 4, l = lane & 15;
#pragma unroll
  for (int i = 0; i < 2; ++i)
#pragma unroll
    for (int j = 0; j < 2; ++j)
#pragma unroll
      for (int r = 0; r < 8; ++r)
        L[(size_t)(wn + j * 16 + l) * 132 + wm + i * 16 + r + 8 * h] = acc[i][j][r];
  __syncthreads();
  {
    int ol = t >> 2, seg = t & 3;          // o-row 0..63, 32 m-values each
    int go = o0 + ol;
    float sc = bns[go];
    float c0f = b_out[go] * sc + bnb[go];  // (v + bias)*scale + shift
    const float* rsrc = x + ((size_t)(b * Cc + go)) * Nn + n0 + seg * 32;
    float* dsto = out + ((size_t)(b * Cc + go)) * Nn + n0 + seg * 32;
    const float* Lr = L + (size_t)ol * 132 + seg * 32;
#pragma unroll
    for (int q = 0; q < 32; q += 4) {
      float4 res = *reinterpret_cast<const float4*>(rsrc + q);
      float4 v;
      v.x = Lr[q + 0] * sc + c0f + res.x;
      v.y = Lr[q + 1] * sc + c0f + res.y;
      v.z = Lr[q + 2] * sc + c0f + res.z;
      v.w = Lr[q + 3] * sc + c0f + res.w;
      *reinterpret_cast<float4*>(dsto + q) = v;
    }
  }
}

extern "C" void kernel_launch(void* const* d_in, const int* in_sizes, int n_in,
                              void* d_out, int out_size, void* d_ws, size_t ws_size,
                              hipStream_t stream) {
  const float* x     = (const float*)d_in[0];
  const float* w_in  = (const float*)d_in[1];
  const float* b_in  = (const float*)d_in[2];
  const float* w_out = (const float*)d_in[3];
  const float* b_out = (const float*)d_in[4];
  const float* gamma = (const float*)d_in[5];
  const float* beta  = (const float*)d_in[6];
  const float* rmean = (const float*)d_in[7];
  const float* rvar  = (const float*)d_in[8];
  const float* bases = (const float*)d_in[9];

  char* ws = (char*)d_ws;
  bf16_t* xf    = (bf16_t*)(ws);                 // B*N*C bf16   = 64 MB
  bf16_t* zbuf  = (bf16_t*)(ws + 67108864);      // B*N*K bf16   =  8 MB
  bf16_t* rec   = (bf16_t*)(ws + 75497472);      // B*N*C bf16   = 64 MB
  bf16_t* mu    = (bf16_t*)(ws + 142606336);     // B*K*C bf16   =  1 MB
  float*  ztx   = (float*) (ws + 143654912);     // B*K*C f32    =  2 MB
  bf16_t* winT  = (bf16_t*)(ws + 145752064);     // C*C bf16     = .5 MB
  bf16_t* woutT = (bf16_t*)(ws + 146276352);     // C*C bf16     = .5 MB
  float*  bns   = (float*) (ws + 146800640);     // C f32
  float*  bnb   = (float*) (ws + 146802688);     // C f32
  float*  out   = (float*)d_out;

  k_prep<<<2048, 256, 0, stream>>>(w_in, w_out, bases, gamma, beta, rmean, rvar,
                                   winT, woutT, mu, bns, bnb);
  k_conv_in<<<dim3(32, 8, 16), 256, 0, stream>>>(x, winT, b_in, xf);
  for (int it = 0; it < 3; ++it) {
    k_z<<<dim3(32, 1, 16), 256, 0, stream>>>(xf, mu, zbuf);
    k_mu_gemm<<<dim3(4, 1, 16), 256, 0, stream>>>(zbuf, xf, ztx);
    k_mu_norm<<<128, 256, 0, stream>>>(ztx, mu);
  }
  k_recon<<<dim3(32, 8, 16), 256, 0, stream>>>(zbuf, mu, rec);
  k_conv_out<<<dim3(32, 8, 16), 256, 0, stream>>>(rec, woutT, b_out, bns, bnb, x, out);
}
